// MultiHeadAttention_6597069766777
// MI455X (gfx1250) — matrix operand order repaired and checked by reading, not compiled, not run on hardware
//
#include <hip/hip_runtime.h>
#include <hip/hip_bf16.h>
#include <math.h>

// Problem constants (match reference)
#define B_   2
#define S_   1024
#define D_   512
#define H_   8
#define DK_  64
#define M_   8192
#define KNN_ 32
#define BH_  (B_ * H_)     // 16
#define N_   (B_ * S_)     // 2048 rows of x
#define NQ_  (B_ * S_)     // 2048 memory-path queries per head

typedef __attribute__((ext_vector_type(16))) __bf16 v16bf;
typedef __attribute__((ext_vector_type(8)))  __bf16 v8bf;
typedef __attribute__((ext_vector_type(8)))  float  v8f;

// ---------- helpers ----------
__device__ inline __bf16 f2bf(float f) {
  return (__bf16)f;   // hardware v_cvt_pk_bf16_f32 (RNE)
}

__device__ inline v16bf cat8(v8bf lo, v8bf hi) {
  return __builtin_shufflevector(lo, hi, 0, 1, 2, 3, 4, 5, 6, 7,
                                 8, 9, 10, 11, 12, 13, 14, 15);
}

__device__ inline v8f wmma_bf16(v16bf a, v16bf b, v8f c) {
  // D = A(16x32 bf16) * B(32x16 bf16) + C(16x16 f32)
  return __builtin_amdgcn_wmma_f32_16x16x32_bf16(false, a, false, b, (short)0, c,
                                                 false, false);
}

// A-fragment (16x32, MxK) from bf16 row-major [ld] source (16B-aligned chunks).
// Contiguous-k order (same as load_bt_bf): lane m = lane&15; lanes<16 hold K 0-15, lanes>=16 K 16-31.
__device__ inline v16bf load_a_bf(const __bf16* p, int ld, int row0, int k0, int lane) {
  int m = lane & 15, kh = lane >> 4;
  const __bf16* r = p + (size_t)(row0 + m) * ld + k0;
  v8bf lo = *(const v8bf*)(r + kh * 16);
  v8bf hi = *(const v8bf*)(r + kh * 16 + 8);
  return cat8(lo, hi);
}

// B-fragment (32x16, KxN) where the source array stores B transposed, i.e. src[n][k]
// row-major with stride ld. Contiguous-k order (same as load_a_bf): lane n = lane&15; lanes<16 hold K 0-15,
// lanes>=16 hold K 16-31 -> one contiguous 32B read per lane (two b128 vectors).
__device__ inline v16bf load_bt_bf(const __bf16* p, int ld, int n0, int k0, int lane) {
  int n = lane & 15, kh = lane >> 4;
  const __bf16* r = p + (size_t)(n0 + n) * ld + k0 + kh * 16;
  v8bf lo = *(const v8bf*)(r);
  v8bf hi = *(const v8bf*)(r + 8);
  return cat8(lo, hi);
}

__device__ inline float redmax16(float v) {  // reduce within 16-lane half
  v = fmaxf(v, __shfl_xor(v, 1));
  v = fmaxf(v, __shfl_xor(v, 2));
  v = fmaxf(v, __shfl_xor(v, 4));
  v = fmaxf(v, __shfl_xor(v, 8));
  return v;
}
__device__ inline float redsum16(float v) {
  v += __shfl_xor(v, 1); v += __shfl_xor(v, 2);
  v += __shfl_xor(v, 4); v += __shfl_xor(v, 8);
  return v;
}
__device__ inline float redsum32(float v) {
  v += __shfl_xor(v, 1); v += __shfl_xor(v, 2); v += __shfl_xor(v, 4);
  v += __shfl_xor(v, 8); v += __shfl_xor(v, 16);
  return v;
}

// ---------- kernel 1a: transpose + bf16-convert weights (Wt[n][k] = W[k][n]) ----------
__global__ void wt_kernel(const float* __restrict__ Wq, const float* __restrict__ Wk,
                          const float* __restrict__ Wv, const float* __restrict__ Wo,
                          __bf16* __restrict__ wtq, __bf16* __restrict__ wtk,
                          __bf16* __restrict__ wtv, __bf16* __restrict__ wto) {
  int t   = blockIdx.x * blockDim.x + threadIdx.x;   // 0..262143
  int mat = blockIdx.y;
  const float* W  = mat == 0 ? Wq : mat == 1 ? Wk : mat == 2 ? Wv : Wo;
  __bf16*      Wt = mat == 0 ? wtq : mat == 1 ? wtk : mat == 2 ? wtv : wto;
  int k = t >> 9, n = t & 511;
  Wt[(size_t)n * D_ + k] = f2bf(W[(size_t)k * D_ + n]);
}

// ---------- kernel 1b: one-time bf16 copy of x (shared by the 3 QKV GEMMs) ----------
__global__ void xcvt_kernel(const float* __restrict__ x, __bf16* __restrict__ xbf) {
  int i = blockIdx.x * blockDim.x + threadIdx.x;     // 0..N_*D_-1
  xbf[i] = f2bf(x[i]);
}

// ---------- kernel 2: fused QKV projection (WMMA GEMM) ----------
// grid (N_/16, D_/64, 3), block 32 (one wave -> 16x64 output tile)
__global__ void qkv_kernel(const __bf16* __restrict__ xbf, const __bf16* __restrict__ wtq,
                           const __bf16* __restrict__ wtk, const __bf16* __restrict__ wtv,
                           const float* __restrict__ bq, const float* __restrict__ bk,
                           const float* __restrict__ bv, float* __restrict__ qf32,
                           __bf16* __restrict__ qbf, __bf16* __restrict__ kbf,
                           __bf16* __restrict__ vt) {
  int lane = threadIdx.x;
  int rowbase = blockIdx.x * 16, colbase = blockIdx.y * 64, which = blockIdx.z;
  const __bf16* Wt   = which == 0 ? wtq : which == 1 ? wtk : wtv;
  const float*  bias = which == 0 ? bq : which == 1 ? bk : bv;
  v8f c[4] = {};
  for (int kb = 0; kb < D_; kb += 32) {
    v16bf a = load_a_bf(xbf, D_, rowbase, kb, lane);
#pragma unroll
    for (int t = 0; t < 4; ++t)
      c[t] = wmma_bf16(a, load_bt_bf(Wt, D_, colbase + t * 16, kb, lane), c[t]);
  }
  int hi = lane >> 4, n_ = lane & 15;
#pragma unroll
  for (int t = 0; t < 4; ++t) {
    int col = colbase + t * 16 + n_;
    float bb = bias[col];
#pragma unroll
    for (int i = 0; i < 8; ++i) {            // C layout: row = i (+8 for upper half)
      int row = rowbase + i + hi * 8;
      float val = c[t][i] + bb;
      int b = row >> 10, s = row & 1023;
      int h = col >> 6, dk = col & 63;
      int bh = b * H_ + h;
      size_t qi = ((size_t)bh * S_ + s) * DK_ + dk;
      if (which == 0) {                       // q: f32 for norms + bf16 pre-scaled 1/sqrt(DK)
        qf32[qi] = val;
        qbf[qi]  = f2bf(val * 0.125f);
      } else if (which == 1) {
        kbf[qi] = f2bf(val);
      } else {                                // v stored transposed [bh][dk][s]
        vt[((size_t)bh * DK_ + dk) * S_ + s] = f2bf(val);
      }
    }
  }
}

// ---------- kernel 3: cosine-normalize q into memory-path layout ----------
// qn[h'][q3][dk], with the reference's reshape(H,b,..) scramble encoded.
__global__ void qn_kernel(const float* __restrict__ qf32, __bf16* __restrict__ qn) {
  int wid = threadIdx.x >> 5, lane = threadIdx.x & 31;
  int rowid = blockIdx.x * 8 + wid;          // 0..16383 = H*NQ
  int Hp = rowid >> 11, q3 = rowid & 2047;
  int g  = Hp * 2 + (q3 >> 10);              // scramble inverse
  int bh = (g & 7) * 2 + (g >> 3);
  int s  = q3 & 1023;
  const float* src = qf32 + ((size_t)bh * S_ + s) * DK_;
  float v0 = src[lane * 2], v1 = src[lane * 2 + 1];
  float ss = redsum32(v0 * v0 + v1 * v1);
  float sc = 1.0f / ((sqrtf(ss) + 1e-8f) * 8.0f);   // fold 1/sqrt(DK)
  __bf16* dst = qn + (size_t)rowid * DK_;
  dst[lane * 2]     = f2bf(v0 * sc);
  dst[lane * 2 + 1] = f2bf(v1 * sc);
}

// ---------- kernel 4: normalize key_memories ----------
__global__ void kn_kernel(const float* __restrict__ km, __bf16* __restrict__ kn) {
  int wid = threadIdx.x >> 5, lane = threadIdx.x & 31;
  int rowid = blockIdx.x * 8 + wid;          // 0..65535 = H*M
  const float* src = km + (size_t)rowid * DK_;
  float v0 = src[lane * 2], v1 = src[lane * 2 + 1];
  float ss = redsum32(v0 * v0 + v1 * v1);
  float sc = 1.0f / (sqrtf(ss) + 1e-8f);
  __bf16* dst = kn + (size_t)rowid * DK_;
  dst[lane * 2]     = f2bf(v0 * sc);
  dst[lane * 2 + 1] = f2bf(v1 * sc);
}

// ---------- kernel 5: causal flash attention, one wave per (bh, 16-query tile) ----------
__global__ void attn_kernel(const __bf16* __restrict__ qbf, const __bf16* __restrict__ kbf,
                            const __bf16* __restrict__ vtb, float* __restrict__ AO) {
  __shared__ __bf16 p_tile[16][32];          // P staged C-layout -> A-layout
  int lane = threadIdx.x;
  int qs = blockIdx.x * 16, bh = blockIdx.y;
  const __bf16* qp = qbf + (size_t)bh * S_ * DK_;
  const __bf16* kp = kbf + (size_t)bh * S_ * DK_;
  const __bf16* vp = vtb + (size_t)bh * DK_ * S_;
  v16bf aq0 = load_a_bf(qp, DK_, qs, 0, lane);
  v16bf aq1 = load_a_bf(qp, DK_, qs, 32, lane);
  v8f o[4] = {};
  float mrun[8], lrun[8];
#pragma unroll
  for (int i = 0; i < 8; ++i) { mrun[i] = -3.0e38f; lrun[i] = 0.f; }
  int hi = lane >> 4, n_ = lane & 15;
  int nkb = (qs + 16 + 31) / 32;             // causal bound in 32-key blocks
  for (int kb = 0; kb < nkb; ++kb) {
    int k0 = kb * 32;
    v8f s0 = {}, s1 = {};
    s0 = wmma_bf16(aq0, load_bt_bf(kp, DK_, k0, 0, lane), s0);
    s0 = wmma_bf16(aq1, load_bt_bf(kp, DK_, k0, 32, lane), s0);
    s1 = wmma_bf16(aq0, load_bt_bf(kp, DK_, k0 + 16, 0, lane), s1);
    s1 = wmma_bf16(aq1, load_bt_bf(kp, DK_, k0 + 16, 32, lane), s1);
#pragma unroll
    for (int i = 0; i < 8; ++i) {            // per-row online softmax (row lives in one half)
      int row = i + hi * 8;
      int qg  = qs + row;
      float e0 = (k0 + n_      <= qg) ? s0[i] : -3.0e38f;
      float e1 = (k0 + 16 + n_ <= qg) ? s1[i] : -3.0e38f;
      float rm = redmax16(fmaxf(e0, e1));
      float mo = mrun[i];
      float mn = fmaxf(mo, rm);
      float p0 = __expf(e0 - mn);
      float p1 = __expf(e1 - mn);
      float rs = redsum16(p0 + p1);
      float scl = __expf(mo - mn);
      lrun[i] = lrun[i] * scl + rs;
      mrun[i] = mn;
#pragma unroll
      for (int t = 0; t < 4; ++t) o[t][i] *= scl;
      p_tile[row][n_]      = f2bf(p0);
      p_tile[row][16 + n_] = f2bf(p1);
    }
    __syncthreads();
    v16bf ap = load_a_bf(&p_tile[0][0], 32, 0, 0, lane);
#pragma unroll
    for (int t = 0; t < 4; ++t) {            // O += P(16x32) * V(32x16) per dk tile
      v16bf bfrag = load_bt_bf(vp, S_, t * 16, k0, lane);
      o[t] = wmma_bf16(ap, bfrag, o[t]);
    }
    __syncthreads();
  }
#pragma unroll
  for (int i = 0; i < 8; ++i) {
    int row = i + hi * 8;
    float inv = 1.0f / lrun[i];
#pragma unroll
    for (int t = 0; t < 4; ++t)
      AO[((size_t)bh * S_ + qs + row) * DK_ + t * 16 + n_] = o[t][i] * inv;
  }
}

// ---------- kernel 6: similarity GEMM + streaming top-32 ----------
// grid (NQ_/16, H_), block 32. Owner-lanes 0..15 hold a per-row top-32 list in LDS.
__global__ void topk_kernel(const __bf16* __restrict__ qn, const __bf16* __restrict__ kn,
                            float* __restrict__ ts, int* __restrict__ ti) {
  __shared__ float stile[16][17];
  __shared__ float ls[16][KNN_];
  __shared__ int   li[16][KNN_];
  int lane = threadIdx.x;
  int q0 = blockIdx.x * 16, h = blockIdx.y;
  const __bf16* qp = qn + (size_t)h * NQ_ * DK_;
  const __bf16* kp = kn + (size_t)h * M_ * DK_;
  v16bf a0 = load_a_bf(qp, DK_, q0, 0, lane);
  v16bf a1 = load_a_bf(qp, DK_, q0, 32, lane);
  int hi = lane >> 4, n_ = lane & 15;
  float minv = 0.f; int minp = 0;
  for (int mb = 0; mb < M_; mb += 16) {
    v8f s = {};
    s = wmma_bf16(a0, load_bt_bf(kp, DK_, mb, 0, lane), s);
    s = wmma_bf16(a1, load_bt_bf(kp, DK_, mb, 32, lane), s);
#pragma unroll
    for (int i = 0; i < 8; ++i) stile[i + hi * 8][n_] = s[i];
    __syncthreads();
    if (lane < 16) {
      int row = lane;
      if (mb == 0) {
#pragma unroll
        for (int j = 0; j < 16; ++j) { ls[row][j] = stile[row][j]; li[row][j] = j; }
      } else if (mb == 16) {
#pragma unroll
        for (int j = 0; j < 16; ++j) { ls[row][16 + j] = stile[row][j]; li[row][16 + j] = 16 + j; }
        minv = ls[row][0]; minp = 0;
#pragma unroll
        for (int j = 1; j < KNN_; ++j)
          if (ls[row][j] < minv) { minv = ls[row][j]; minp = j; }
      } else {
        for (int j = 0; j < 16; ++j) {       // threshold filter; rare replacements rescan
          float sc = stile[row][j];
          if (sc > minv) {
            ls[row][minp] = sc; li[row][minp] = mb + j;
            minv = ls[row][0]; minp = 0;
#pragma unroll
            for (int jj = 1; jj < KNN_; ++jj)
              if (ls[row][jj] < minv) { minv = ls[row][jj]; minp = jj; }
          }
        }
      }
    }
    __syncthreads();
  }
  if (lane < 16) {
    int row = lane;
    size_t base = ((size_t)h * NQ_ + q0 + row) * KNN_;
    for (int j = 0; j < KNN_; ++j) { ts[base + j] = ls[row][j]; ti[base + j] = li[row][j]; }
  }
}

// ---------- kernel 7: gather values, weighted sum, sigmoid gate, combine ----------
// Writes the pre-Wo activation directly in bf16 so the output GEMM never converts.
__global__ void combine_kernel(const float* __restrict__ ts, const int* __restrict__ ti,
                               const float* __restrict__ vm, const float* __restrict__ AO,
                               const float* __restrict__ mg, __bf16* __restrict__ X2bf) {
  int wid = threadIdx.x >> 5, lane = threadIdx.x & 31;
  int rowid = blockIdx.x * 8 + wid;          // 0..16383
  int Hp = rowid >> 11, q3 = rowid & 2047;
  int Bp = q3 >> 10, s = q3 & 1023;
  int bh_attn = Hp * 2 + Bp;                 // attn tensor coords after reference scramble
  const float* tsp = ts + (size_t)rowid * KNN_;
  const int*   tip = ti + (size_t)rowid * KNN_;
  const float* vmb = vm + (size_t)Hp * M_ * DK_;
  int dk0 = lane * 2;
  float a0 = 0.f, a1 = 0.f;
  for (int j = 0; j < KNN_; ++j) {
    float sc = tsp[j]; int idx = tip[j];
    const float* vr = vmb + (size_t)idx * DK_;
    a0 += sc * vr[dk0]; a1 += sc * vr[dk0 + 1];
  }
  float g = 1.0f / (1.0f + __expf(-mg[Hp]));
  const float* aop = AO + ((size_t)bh_attn * S_ + s) * DK_;
  float o0 = g * a0 + (1.0f - g) * aop[dk0];
  float o1 = g * a1 + (1.0f - g) * aop[dk0 + 1];
  __bf16* dst = X2bf + ((size_t)(Bp * S_ + s)) * D_ + Hp * DK_;
  dst[dk0]     = f2bf(o0);
  dst[dk0 + 1] = f2bf(o1);
}

// ---------- kernel 8: output projection ----------
__global__ void outproj_kernel(const __bf16* __restrict__ X2bf, const __bf16* __restrict__ wto,
                               const float* __restrict__ bo, float* __restrict__ out) {
  int lane = threadIdx.x;
  int rowbase = blockIdx.x * 16, colbase = blockIdx.y * 64;
  v8f c[4] = {};
  for (int kb = 0; kb < D_; kb += 32) {
    v16bf a = load_a_bf(X2bf, D_, rowbase, kb, lane);
#pragma unroll
    for (int t = 0; t < 4; ++t)
      c[t] = wmma_bf16(a, load_bt_bf(wto, D_, colbase + t * 16, kb, lane), c[t]);
  }
  int hi = lane >> 4, n_ = lane & 15;
#pragma unroll
  for (int t = 0; t < 4; ++t) {
    int col = colbase + t * 16 + n_;
    float bb = bo[col];
#pragma unroll
    for (int i = 0; i < 8; ++i) {
      int row = rowbase + i + hi * 8;
      out[(size_t)row * D_ + col] = c[t][i] + bb;
    }
  }
}

// ---------- launch ----------
extern "C" void kernel_launch(void* const* d_in, const int* in_sizes, int n_in,
                              void* d_out, int out_size, void* d_ws, size_t ws_size,
                              hipStream_t stream) {
  (void)in_sizes; (void)n_in; (void)out_size; (void)ws_size;
  const float* x  = (const float*)d_in[0];
  const float* Wq = (const float*)d_in[1];
  const float* bq = (const float*)d_in[2];
  const float* Wk = (const float*)d_in[3];
  const float* bk = (const float*)d_in[4];
  const float* Wv = (const float*)d_in[5];
  const float* bv = (const float*)d_in[6];
  const float* Wo = (const float*)d_in[7];
  const float* bo = (const float*)d_in[8];
  const float* km = (const float*)d_in[9];
  const float* vm = (const float*)d_in[10];
  const float* mg = (const float*)d_in[11];
  float* out = (float*)d_out;

  char* ws = (char*)d_ws;
  size_t off = 0;
  auto alloc = [&](size_t bytes) -> void* {
    void* p = ws + off;
    off += (bytes + 255) & ~(size_t)255;
    return p;
  };
  float*  qf32 = (float*) alloc((size_t)BH_ * S_ * DK_ * 4);         // 4 MB
  __bf16* qbf  = (__bf16*)alloc((size_t)BH_ * S_ * DK_ * 2);         // 2 MB
  __bf16* kbf  = (__bf16*)alloc((size_t)BH_ * S_ * DK_ * 2);         // 2 MB
  __bf16* vt   = (__bf16*)alloc((size_t)BH_ * DK_ * S_ * 2);         // 2 MB
  float*  AO   = (float*) alloc((size_t)BH_ * S_ * DK_ * 4);         // 4 MB
  __bf16* qn   = (__bf16*)alloc((size_t)H_ * NQ_ * DK_ * 2);         // 2 MB
  __bf16* kn   = (__bf16*)alloc((size_t)H_ * M_ * DK_ * 2);          // 8 MB
  float*  ts   = (float*) alloc((size_t)H_ * NQ_ * KNN_ * 4);        // 2 MB
  int*    ti   = (int*)   alloc((size_t)H_ * NQ_ * KNN_ * 4);        // 2 MB
  __bf16* X2bf = (__bf16*)alloc((size_t)N_ * D_ * 2);                // 2 MB
  __bf16* xbf  = (__bf16*)alloc((size_t)N_ * D_ * 2);                // 2 MB
  __bf16* wtq  = (__bf16*)alloc((size_t)D_ * D_ * 2);
  __bf16* wtk  = (__bf16*)alloc((size_t)D_ * D_ * 2);
  __bf16* wtv  = (__bf16*)alloc((size_t)D_ * D_ * 2);
  __bf16* wto  = (__bf16*)alloc((size_t)D_ * D_ * 2);                // +2 MB, ~34 MB total

  wt_kernel<<<dim3(1024, 4), 256, 0, stream>>>(Wq, Wk, Wv, Wo, wtq, wtk, wtv, wto);
  xcvt_kernel<<<(N_ * D_) / 256, 256, 0, stream>>>(x, xbf);
  qkv_kernel<<<dim3(N_ / 16, D_ / 64, 3), 32, 0, stream>>>(
      xbf, wtq, wtk, wtv, bq, bk, bv, qf32, qbf, kbf, vt);
  qn_kernel<<<(H_ * NQ_) / 8, 256, 0, stream>>>(qf32, qn);
  kn_kernel<<<(H_ * M_) / 8, 256, 0, stream>>>(km, kn);
  attn_kernel<<<dim3(S_ / 16, BH_), 32, 0, stream>>>(qbf, kbf, vt, AO);
  topk_kernel<<<dim3(NQ_ / 16, H_), 32, 0, stream>>>(qn, kn, ts, ti);
  combine_kernel<<<(H_ * NQ_) / 8, 256, 0, stream>>>(ts, ti, vm, AO, mg, X2bf);
  outproj_kernel<<<dim3(N_ / 16, D_ / 64), 32, 0, stream>>>(X2bf, wto, bo, out);
}
